// MultiHeadAttention_71554155152149
// MI455X (gfx1250) — matrix operand order repaired and checked by reading, not compiled, not run on hardware
//
#include <hip/hip_runtime.h>

// ---------------- problem constants ----------------
constexpr int kB  = 4;
constexpr int kT  = 2048;
constexpr int kD  = 1024;
constexpr int kH  = 16;
constexpr int kHS = 64;

// ---------------- types ----------------
typedef __bf16 bf16;
typedef __attribute__((ext_vector_type(16))) __bf16 v16bf;
typedef __attribute__((ext_vector_type(8)))  __bf16 v8bf;
typedef __attribute__((ext_vector_type(8)))  float  v8f;
typedef __attribute__((ext_vector_type(4)))  int    vint4;
typedef __attribute__((ext_vector_type(4)))  unsigned int v4u;
typedef __attribute__((ext_vector_type(8)))  int    v8i;
typedef __attribute__((ext_vector_type(4)))  int    v4i;

#if defined(__gfx1250__) && __has_builtin(__builtin_amdgcn_tensor_load_to_lds)
#define HAS_TDM 1
#else
#define HAS_TDM 0
#endif

__device__ __forceinline__ bf16 to_bf16(float f) {
  // RNE fp32 -> bf16 via bit math
  unsigned u = __builtin_bit_cast(unsigned, f);
  unsigned r = u + 0x7fffu + ((u >> 16) & 1u);
  unsigned short h = (unsigned short)(r >> 16);
  return __builtin_bit_cast(bf16, h);
}

// Async 16-byte global -> LDS copy (ASYNCcnt path on gfx1250).
__device__ __forceinline__ void async_copy_b128(const void* g, void* l) {
#if defined(__gfx1250__) && __has_builtin(__builtin_amdgcn_global_load_async_to_lds_b128)
  __builtin_amdgcn_global_load_async_to_lds_b128(
      (__attribute__((address_space(1))) vint4*)(vint4*)g,
      (__attribute__((address_space(3))) vint4*)(vint4*)l,
      0, 0);
#else
  *(uint4*)l = *(const uint4*)g;
#endif
}

__device__ __forceinline__ void wait_async() {
#if defined(__gfx1250__) && __has_builtin(__builtin_amdgcn_s_wait_asynccnt)
  __builtin_amdgcn_s_wait_asynccnt(0);
#elif defined(__gfx1250__)
  asm volatile("s_wait_asynccnt 0x0" ::: "memory");
#endif
}

// LDS byte offset of a generic pointer to __shared__ data.
__device__ __forceinline__ unsigned lds_off(const void* p) {
  return (unsigned)(size_t)(__attribute__((address_space(3))) const void*)p;
}

// TDM 2D tile load: tile_d1 rows of tile_d0 bf16 elements, row stride stride0
// (all in elements); writes the tile contiguously to LDS.  ISA 08_async_tensor §8.
__device__ __forceinline__ void tdm_load_2d(const void* gsrc, void* ldst,
                                            unsigned tile_d0, unsigned tile_d1,
                                            unsigned stride0,
                                            unsigned tensor_d0, unsigned tensor_d1) {
#if HAS_TDM
  unsigned long long ga = (unsigned long long)(size_t)gsrc;
  v4u g0;
  g0.x = 1u;                                                   // D# count=1, user mode
  g0.y = lds_off(ldst);                                        // lds_addr (bytes)
  g0.z = (unsigned)ga;                                         // global_addr[31:0]
  g0.w = (unsigned)((ga >> 32) & 0x01FFFFFFull) | (2u << 30);  // addr[56:32] | type=2
  v8i g1;
  g1[0] = (int)(1u << 16);                                     // data_size=1 (2 bytes)
  g1[1] = (int)((tensor_d0 & 0xFFFFu) << 16);                  // tensor_dim0[15:0]
  g1[2] = (int)((tensor_d0 >> 16) | ((tensor_d1 & 0xFFFFu) << 16));
  g1[3] = (int)((tensor_d1 >> 16) | (tile_d0 << 16));          // tile_dim0
  g1[4] = (int)tile_d1;                                        // tile_dim1, tile_dim2=0
  g1[5] = (int)stride0;                                        // tensor_dim0_stride lo32
  g1[6] = 0;
  g1[7] = 0;
  v4i g2 = {0, 0, 0, 0};
  v4i g3 = {0, 0, 0, 0};
  v8i g4 = {};                                                 // 6-arg form: extra group
  __builtin_amdgcn_tensor_load_to_lds(g0, g1, g2, g3, g4, 0);
#else
  (void)gsrc; (void)ldst; (void)tile_d0; (void)tile_d1;
  (void)stride0; (void)tensor_d0; (void)tensor_d1;
#endif
}

__device__ __forceinline__ void wait_tensor0() {
#if HAS_TDM
  __builtin_amdgcn_s_wait_tensorcnt(0);
#endif
}
__device__ __forceinline__ void wait_tensor_le2() {
#if HAS_TDM
  __builtin_amdgcn_s_wait_tensorcnt(2);
#endif
}

// A fragment (16x32, row-major tile): lane<16 -> row=lane, k in {0..7,16..23};
// lane>=16 -> row=lane-16, k in {8..15,24..31}.  (ISA 7.12.2)
__device__ __forceinline__ v16bf frag_a_row(const bf16* tile, int ld) {
  const int lane = threadIdx.x & 31;
  const int m  = lane & 15;
  const int kb = (lane & 16) >> 1;          // 0 or 8
  const bf16* p = tile + m * ld + kb;
  v8bf lo = *(const v8bf*)(p);
  v8bf hi = *(const v8bf*)(p + 16);
  return __builtin_shufflevector(lo, hi, 0,1,2,3,4,5,6,7,8,9,10,11,12,13,14,15);
}

// B fragment (32x16), stored column-contiguous: element (k,n) at tile[n*ld+k].
__device__ __forceinline__ v16bf frag_b_col(const bf16* tile, int ld) {
  const int lane = threadIdx.x & 31;
  const int n  = lane & 15;
  const int kh = (lane & 16) >> 1;          // 0 or 8
  const bf16* p = tile + n * ld + kh;
  v8bf lo = *(const v8bf*)(p);
  v8bf hi = *(const v8bf*)(p + 16);
  return __builtin_shufflevector(lo, hi, 0,1,2,3,4,5,6,7,8,9,10,11,12,13,14,15);
}

__device__ __forceinline__ v8f wmma_bf16(v16bf a, v16bf b, v8f c) {
  return __builtin_amdgcn_wmma_f32_16x16x32_bf16(
      false, a, false, b, (short)0, c, false, false);
}

// ---------------- kernel 0: fp32 -> bf16 weight conversion ----------------
__global__ __launch_bounds__(256) void cvt_bf16_kernel(const float* __restrict__ src,
                                                       bf16* __restrict__ dst, int n) {
  for (int i = (blockIdx.x * 256 + threadIdx.x) * 4; i < n; i += gridDim.x * 256 * 4) {
    float4 f = *(const float4*)(src + i);
    dst[i + 0] = to_bf16(f.x); dst[i + 1] = to_bf16(f.y);
    dst[i + 2] = to_bf16(f.z); dst[i + 3] = to_bf16(f.w);
  }
}

// ---------------- kernel 1: QKV projection (per head) ----------------
// grid: (T/64, B*H, 3).  Q,K stored [bh][t][e]; V stored TRANSPOSED [bh][e][t].
__global__ __launch_bounds__(256) void qkv_proj_kernel(
    const float* __restrict__ x,
    const bf16* __restrict__ Wq, const bf16* __restrict__ Wk, const bf16* __restrict__ Wv,
    bf16* __restrict__ Q, bf16* __restrict__ K, bf16* __restrict__ V) {
  const int tid  = threadIdx.x;
  const int wave = tid >> 5;
  const int i0   = blockIdx.x * 64;
  const int bh   = blockIdx.y;
  const int z    = blockIdx.z;

  const bf16* W = (z == 0) ? Wq : (z == 1) ? Wk : Wv;
  bf16*       O = (z == 0) ? Q  : (z == 1) ? K  : V;
  const float scale = (z == 0) ? 0.125f : 1.0f;   // HS^-0.5 folded into Q

  __shared__ __align__(32) bf16 Xs[64 * 64];      // [row][k]
  __shared__ __align__(32) bf16 Ws[64 * 64];      // [n][k] (transposed)

  const float* xg = x + (size_t)(bh / kH) * kT * kD + (size_t)i0 * kD;
  const bf16*  Wg = W + (size_t)(bh % kH) * kD * kHS;

  const int rb = wave & 3;
  const int cb = (wave >> 2) * 2;
  v8f acc0 = {}, acc1 = {};

  for (int kk = 0; kk < kD; kk += 64) {
    for (int i = tid * 4; i < 64 * 64; i += 1024) {      // X 64x64 fp32->bf16
      int r = i >> 6, c = i & 63;
      float4 f = *(const float4*)(xg + (size_t)r * kD + kk + c);
      Xs[i + 0] = to_bf16(f.x); Xs[i + 1] = to_bf16(f.y);
      Xs[i + 2] = to_bf16(f.z); Xs[i + 3] = to_bf16(f.w);
    }
    for (int i = tid; i < 64 * 64; i += 256) {           // W 64x64 -> [n][k]
      int k = i >> 6, n = i & 63;
      Ws[n * 64 + k] = Wg[(size_t)(kk + k) * kHS + n];
    }
    if (kk + 64 < kD)                                    // prefetch next X slab
      __builtin_prefetch(xg + (size_t)(tid >> 2) * kD + kk + 64 + (tid & 3) * 16, 0, 1);
    __syncthreads();
#pragma unroll
    for (int kkk = 0; kkk < 64; kkk += 32) {
      v16bf a  = frag_a_row(Xs + rb * 16 * 64 + kkk, 64);
      v16bf b0 = frag_b_col(Ws + (cb * 16) * 64 + kkk, 64);
      v16bf b1 = frag_b_col(Ws + ((cb + 1) * 16) * 64 + kkk, 64);
      acc0 = wmma_bf16(a, b0, acc0);
      acc1 = wmma_bf16(a, b1, acc1);
    }
    __syncthreads();
  }

  // repack through LDS -> b128 coalesced stores (V transposed)
  const int lane = tid & 31;
  const int n  = lane & 15;
  const int mb = (lane & 16) >> 1;
#pragma unroll
  for (int v = 0; v < 8; ++v) {
    int row = rb * 16 + mb + v;
    int c0 = cb * 16 + n, c1 = (cb + 1) * 16 + n;
    bf16 e0 = to_bf16(acc0[v] * scale);
    bf16 e1 = to_bf16(acc1[v] * scale);
    if (z < 2) { Xs[row * 64 + c0] = e0; Xs[row * 64 + c1] = e1; }
    else       { Xs[c0 * 64 + row] = e0; Xs[c1 * 64 + row] = e1; }  // [e][t]
  }
  __syncthreads();
  bf16* base = (z < 2) ? O + ((size_t)bh * kT + i0) * kHS
                       : O + (size_t)bh * kHS * kT + i0;
  const size_t stride = (z < 2) ? kHS : kT;
  for (int i = tid * 8; i < 64 * 64; i += 2048) {
    int r = i >> 6, c = i & 63;
    *(uint4*)(base + (size_t)r * stride + c) = *(const uint4*)(Xs + i);
  }
}

// ---------------- kernel 2: flash attention (causal, online softmax) ----------------
// grid: (T/64, B*H).  V is pre-transposed [bh][e][t].
// K/V tiles staged by the Tensor Data Mover, double-buffered and pipelined:
// tile j+1 is issued before waiting on tile j (same-wave TDM ops complete in
// order, so s_wait_tensorcnt<=2 guarantees the current tile has landed).
__global__ __launch_bounds__(256) void attn_kernel(
    const bf16* __restrict__ Q, const bf16* __restrict__ K, const bf16* __restrict__ Vt_g,
    bf16* __restrict__ Obf) {
  const int tid  = threadIdx.x;
  const int wave = tid >> 5;
  const int lane = tid & 31;
  const int i0   = blockIdx.x * 64;
  const int bh   = blockIdx.y;
  const int b    = bh / kH, h = bh % kH;

  __shared__ __align__(32) bf16  Qs[64 * 64];        // [q][e]
  __shared__ __align__(32) bf16  Ks[2 * 64 * 64];    // [key][e], double-buffered
  __shared__ __align__(32) bf16  Vs[2 * 64 * 64];    // [e][key], double-buffered
  __shared__ __align__(32) float Ss[64 * 64];        // scores f32 / epilogue staging
  __shared__ __align__(32) bf16  Ps[64 * 64];        // probs bf16
  __shared__ float mrow[64], lrow[64], crow[64];
  __shared__ float pm[256], pl[256];

  const bf16* Qg = Q    + ((size_t)bh * kT + i0) * kHS;
  const bf16* Kg = K    + (size_t)bh * kT * kHS;
  const bf16* Vg = Vt_g + (size_t)bh * kHS * kT;

#if HAS_TDM
  if (wave == 0) {                               // issue Q + first K/V tiles
    tdm_load_2d(Qg, Qs, 64, 64, 64, 64, kT);
    tdm_load_2d(Kg, Ks, 64, 64, 64, 64, kT);
    tdm_load_2d(Vg, Vs, 64, 64, kT, kT, 64);
  }
#else
  for (int i = tid * 8; i < 64 * 64; i += 2048)
    async_copy_b128(Qg + i, Qs + i);
#endif
  if (tid < 64) { mrow[tid] = -1e30f; lrow[tid] = 0.0f; }
#if !HAS_TDM
  wait_async();
#endif
  __syncthreads();

  const int rb = wave & 3;
  const int cb = (wave >> 2) * 2;
  v8f o0 = {}, o1 = {};

  for (int j0 = 0; j0 <= i0; j0 += 64) {
    const int buf = (j0 >> 6) & 1;
    bf16* Ksb = Ks + buf * 4096;
    bf16* Vsb = Vs + buf * 4096;
#if HAS_TDM
    if (wave == 0) {
      if (j0 + 64 <= i0) {                       // pipeline next tile, then wait current
        tdm_load_2d(Kg + (size_t)(j0 + 64) * kHS, Ks + (buf ^ 1) * 4096, 64, 64, 64, 64, kT);
        tdm_load_2d(Vg + j0 + 64,                 Vs + (buf ^ 1) * 4096, 64, 64, kT, kT, 64);
        wait_tensor_le2();
      } else {
        wait_tensor0();
      }
    }
#else
    const bf16* kg = Kg + (size_t)j0 * kHS;
    for (int i = tid * 8; i < 64 * 64; i += 2048)
      async_copy_b128(kg + i, Ksb + i);
    for (int i = tid * 8; i < 64 * 64; i += 2048) {
      int r = i >> 6, c = i & 63;                // r = e, c = key
      async_copy_b128(Vg + (size_t)r * kT + j0 + c, Vsb + i);
    }
    wait_async();
#endif
    __syncthreads();

    // ---- S = Q @ K^T (scale folded into Q) ----
    v8f s0 = {}, s1 = {};
#pragma unroll
    for (int kkk = 0; kkk < 64; kkk += 32) {
      v16bf a  = frag_a_row(Qs + rb * 16 * 64 + kkk, 64);
      v16bf b0 = frag_b_col(Ksb + (cb * 16) * 64 + kkk, 64);
      v16bf b1 = frag_b_col(Ksb + ((cb + 1) * 16) * 64 + kkk, 64);
      s0 = wmma_bf16(a, b0, s0);
      s1 = wmma_bf16(a, b1, s1);
    }
    const int n  = lane & 15;
    const int mb = (lane & 16) >> 1;
#pragma unroll
    for (int v = 0; v < 8; ++v) {
      int row = rb * 16 + mb + v;
      Ss[row * 64 + cb * 16 + n]       = s0[v];
      Ss[row * 64 + (cb + 1) * 16 + n] = s1[v];
    }
    __syncthreads();

    // ---- online softmax: 4 threads per row ----
    {
      int row = tid >> 2, seg = tid & 3;
      int kmax = (i0 + row) - j0 + 1; if (kmax > 64) kmax = 64;  // causal
      float mloc = -1e30f;
      for (int c = seg * 16; c < seg * 16 + 16; ++c)
        if (c < kmax) mloc = fmaxf(mloc, Ss[row * 64 + c]);
      pm[tid] = mloc;
    }
    __syncthreads();
    if (tid < 64) {
      float m_old = mrow[tid];
      float m_new = fmaxf(m_old, fmaxf(fmaxf(pm[tid * 4 + 0], pm[tid * 4 + 1]),
                                       fmaxf(pm[tid * 4 + 2], pm[tid * 4 + 3])));
      crow[tid] = exp2f((m_old - m_new) * 1.44269504f);
      mrow[tid] = m_new;
    }
    __syncthreads();
    {
      int row = tid >> 2, seg = tid & 3;
      int kmax = (i0 + row) - j0 + 1; if (kmax > 64) kmax = 64;
      float m_new = mrow[row];
      float s = 0.0f;
      for (int c = seg * 16; c < seg * 16 + 16; ++c) {
        float p = (c < kmax) ? exp2f((Ss[row * 64 + c] - m_new) * 1.44269504f) : 0.0f;
        Ps[row * 64 + c] = to_bf16(p);
        s += p;
      }
      pl[tid] = s;
    }
    __syncthreads();
    if (tid < 64)
      lrow[tid] = lrow[tid] * crow[tid] +
                  pl[tid * 4 + 0] + pl[tid * 4 + 1] + pl[tid * 4 + 2] + pl[tid * 4 + 3];

    // ---- rescale running O, accumulate P @ V ----
#pragma unroll
    for (int v = 0; v < 8; ++v) {
      int row = rb * 16 + mb + v;
      float c = crow[row];
      o0[v] *= c; o1[v] *= c;
    }
#pragma unroll
    for (int kkk = 0; kkk < 64; kkk += 32) {
      v16bf a  = frag_a_row(Ps + rb * 16 * 64 + kkk, 64);
      v16bf b0 = frag_b_col(Vsb + (cb * 16) * 64 + kkk, 64);
      v16bf b1 = frag_b_col(Vsb + ((cb + 1) * 16) * 64 + kkk, 64);
      o0 = wmma_bf16(a, b0, o0);
      o1 = wmma_bf16(a, b1, o1);
    }
    __syncthreads();
  }

  // ---- epilogue: /l, repack via LDS, b128 stores, layout [b][t][h*HS+e] ----
  const int n  = lane & 15;
  const int mb = (lane & 16) >> 1;
  bf16* Ost = (bf16*)Ss;
#pragma unroll
  for (int v = 0; v < 8; ++v) {
    int row = rb * 16 + mb + v;
    float inv = 1.0f / lrow[row];
    Ost[row * 64 + cb * 16 + n]       = to_bf16(o0[v] * inv);
    Ost[row * 64 + (cb + 1) * 16 + n] = to_bf16(o1[v] * inv);
  }
  __syncthreads();
  bf16* Og = Obf + ((size_t)b * kT + i0) * kD + h * kHS;
  for (int i = tid * 8; i < 64 * 64; i += 2048) {
    int r = i >> 6, c = i & 63;
    *(uint4*)(Og + (size_t)r * kD + c) = *(const uint4*)(Ost + i);
  }
}

// ---------------- kernel 3: output projection + bias ----------------
// grid: (D/64, B*T/64). out[8192,1024] = A[8192,1024] @ Wo[1024,1024] + bo
__global__ __launch_bounds__(256) void out_proj_kernel(
    const bf16* __restrict__ A, const bf16* __restrict__ Wo,
    const float* __restrict__ bo, float* __restrict__ out) {
  const int tid  = threadIdx.x;
  const int wave = tid >> 5;
  const int n0   = blockIdx.x * 64;
  const int i0   = blockIdx.y * 64;

  __shared__ __align__(32) bf16 As[64 * 64];   // [row][k]
  __shared__ __align__(32) bf16 Ws[64 * 64];   // [n][k]

  const int rb = wave & 3;
  const int cb = (wave >> 2) * 2;
  v8f acc0 = {}, acc1 = {};

  for (int kk = 0; kk < kD; kk += 64) {
    for (int i = tid * 8; i < 64 * 64; i += 2048) {
      int r = i >> 6, c = i & 63;
      async_copy_b128(A + (size_t)(i0 + r) * kD + kk + c, As + i);
    }
    for (int i = tid; i < 64 * 64; i += 256) {
      int k = i >> 6, n = i & 63;
      Ws[n * 64 + k] = Wo[(size_t)(kk + k) * kD + n0 + n];
    }
    if (kk + 64 < kD)                          // prefetch next Wo slab
      __builtin_prefetch(Wo + (size_t)(kk + 64 + (tid & 63)) * kD + n0 + (tid >> 6) * 32, 0, 1);
    wait_async();
    __syncthreads();
#pragma unroll
    for (int kkk = 0; kkk < 64; kkk += 32) {
      v16bf a  = frag_a_row(As + rb * 16 * 64 + kkk, 64);
      v16bf b0 = frag_b_col(Ws + (cb * 16) * 64 + kkk, 64);
      v16bf b1 = frag_b_col(Ws + ((cb + 1) * 16) * 64 + kkk, 64);
      acc0 = wmma_bf16(a, b0, acc0);
      acc1 = wmma_bf16(a, b1, acc1);
    }
    __syncthreads();
  }

  const int lane = tid & 31;
  const int n  = lane & 15;
  const int mb = (lane & 16) >> 1;
#pragma unroll
  for (int v = 0; v < 8; ++v) {
    int row = i0 + rb * 16 + mb + v;
    int c0 = n0 + cb * 16 + n;
    int c1 = n0 + (cb + 1) * 16 + n;
    out[(size_t)row * kD + c0] = acc0[v] + bo[c0];
    out[(size_t)row * kD + c1] = acc1[v] + bo[c1];
  }
}

// ---------------- launcher ----------------
extern "C" void kernel_launch(void* const* d_in, const int* in_sizes, int n_in,
                              void* d_out, int out_size, void* d_ws, size_t ws_size,
                              hipStream_t stream) {
  (void)in_sizes; (void)n_in; (void)out_size; (void)ws_size;
  const float* x  = (const float*)d_in[0];
  const float* Wq = (const float*)d_in[1];
  const float* Wk = (const float*)d_in[2];
  const float* Wv = (const float*)d_in[3];
  const float* Wo = (const float*)d_in[4];
  const float* bo = (const float*)d_in[5];
  float* out = (float*)d_out;

  const size_t nW   = (size_t)kH * kD * kHS;        // 1M
  const size_t nWo  = (size_t)kD * kD;              // 1M
  const size_t nQKV = (size_t)kB * kH * kT * kHS;   // 8M

  bf16* Wq_bf = (bf16*)d_ws;
  bf16* Wk_bf = Wq_bf + nW;
  bf16* Wv_bf = Wk_bf + nW;
  bf16* Wo_bf = Wv_bf + nW;
  bf16* Qb    = Wo_bf + nWo;
  bf16* Kb    = Qb + nQKV;
  bf16* Vb    = Kb + nQKV;    // transposed per head: [bh][e][t]
  bf16* Ob    = Vb + nQKV;
  // total 72 MB of d_ws

  cvt_bf16_kernel<<<512, 256, 0, stream>>>(Wq, Wq_bf, (int)nW);
  cvt_bf16_kernel<<<512, 256, 0, stream>>>(Wk, Wk_bf, (int)nW);
  cvt_bf16_kernel<<<512, 256, 0, stream>>>(Wv, Wv_bf, (int)nW);
  cvt_bf16_kernel<<<512, 256, 0, stream>>>(Wo, Wo_bf, (int)nWo);

  qkv_proj_kernel<<<dim3(kT / 64, kB * kH, 3), 256, 0, stream>>>(
      x, Wq_bf, Wk_bf, Wv_bf, Qb, Kb, Vb);

  attn_kernel<<<dim3(kT / 64, kB * kH), 256, 0, stream>>>(Qb, Kb, Vb, Ob);

  out_proj_kernel<<<dim3(kD / 64, (kB * kT) / 64), 256, 0, stream>>>(
      Ob, Wo_bf, bo, out);
}
